// MGACMDAAdapter_67654324847551
// MI455X (gfx1250) — matrix operand order repaired and checked by reading, not compiled, not run on hardware
//
#include <hip/hip_runtime.h>
#include <hip/hip_bf16.h>

// ---------------------------------------------------------------------------
// MGACMDAAdapter on MI455X (gfx1250, wave32, WMMA).
// Compute-dominant path: 16 masked-softmax attention passes + GEMMs, all done
// with v_wmma_f32_16x16x32_bf16 (f32 accumulate). Attention is flash-style
// (online softmax) so the 4096x4096 score matrices are never materialized:
// with everything L2-resident the kernel is WMMA-bound, not HBM-bound.
// GEMM waves are register-blocked 16x32 (A fragment reused across 2 WMMAs).
// ---------------------------------------------------------------------------

#define N_NODES 4096
#define IN_DIM  512
#define H1      256
#define H2      128
#define HEADS   4
#define DECD    512
#define N_EDGES 131072
#define B_PAIRS 4096
#define NEGV    (-9e15f)

typedef unsigned int   u32;
typedef unsigned short u16;
typedef __attribute__((ext_vector_type(16))) __bf16 v16bf;
typedef __attribute__((ext_vector_type(8)))  float  v8f;

union Frag16 { v16bf v; u32 u[8]; };

__device__ __forceinline__ u16 f2bf(float f) {
  u32 x = __float_as_uint(f);
  x += 0x7FFFu + ((x >> 16) & 1u);        // round-to-nearest-even
  return (u16)(x >> 16);
}
__device__ __forceinline__ float eluf(float v)     { return v > 0.f ? v : (__expf(v) - 1.f); }
__device__ __forceinline__ float sigm(float v)     { return 1.f / (1.f + __expf(-v)); }

// -------------------------------- utility kernels --------------------------

__global__ void k_zero32(u32* p, int n) {
  int i = blockIdx.x * blockDim.x + threadIdx.x;
  if (i < n) p[i] = 0u;
}

__global__ void k_mask(const int* __restrict__ e, u32* __restrict__ mb) {
  int i = blockIdx.x * blockDim.x + threadIdx.x;
  if (i < N_EDGES) {
    int r = e[i], c = e[N_EDGES + i];
    atomicOr(&mb[(size_t)r * (N_NODES >> 5) + (c >> 5)], 1u << (c & 31));
  }
}

__global__ void k_conv(const float* __restrict__ s, u16* __restrict__ d, int n) {
  int i = blockIdx.x * blockDim.x + threadIdx.x;
  if (i < n) d[i] = f2bf(s[i]);
}

// dst[n*K + k] = bf16(src[k*Nc + n])  : f32 [K,Nc] -> bf16 [Nc,K] (transposed)
__global__ void k_convT(const float* __restrict__ s, u16* __restrict__ d, int K, int Nc) {
  int i = blockIdx.x * blockDim.x + threadIdx.x;
  if (i < K * Nc) {
    int k = i / Nc, n = i % Nc;
    d[(size_t)n * K + k] = f2bf(s[i]);
  }
}

// src[h,n] = Wh[h,n,:] . a[h,:F] ; dst[h,n] = Wh[h,n,:] . a[h,F:2F]
__global__ void k_srcdst(const float* __restrict__ Wh, const float* __restrict__ av,
                         float* __restrict__ srcv, float* __restrict__ dstv, int F) {
  __shared__ float r1[128], r2[128];
  int hn = blockIdx.x;
  int h = hn / N_NODES, n = hn % N_NODES;
  int t = threadIdx.x;
  const float* w = Wh + ((size_t)h * N_NODES + n) * F;
  const float* ap = av + (size_t)h * 2 * F;
  float s = 0.f, d2 = 0.f;
  for (int f = t; f < F; f += 128) { float x = w[f]; s += x * ap[f]; d2 += x * ap[F + f]; }
  r1[t] = s; r2[t] = d2; __syncthreads();
  for (int o = 64; o; o >>= 1) { if (t < o) { r1[t] += r1[t + o]; r2[t] += r2[t + o]; } __syncthreads(); }
  if (!t) { srcv[(size_t)h * N_NODES + n] = r1[0]; dstv[(size_t)h * N_NODES + n] = r2[0]; }
}

// -------------------------------- bf16 WMMA GEMM ---------------------------
// C[M,Nc] = A[M,K](bf16) x BT[Nc,K](bf16, pre-transposed), f32 accumulate.
// Block = 4 waves; each wave owns a 16x32 C tile (A fragment reused across
// two WMMAs per K-step); block covers 16x128.
// modes: 0 store, 1 elu, 2 relu(addsrc+acc) [fused lin+res epilogue], 3 relu(acc+bias)
__global__ void k_gemm(const u16* __restrict__ A, const u16* __restrict__ BT,
                       float* __restrict__ C, int M, int Nc, int K,
                       int mode, const float* __restrict__ bias,
                       const float* __restrict__ addsrc, u16* __restrict__ Cbf) {
  const int wid = threadIdx.x >> 5;
  const int l   = threadIdx.x & 31;
  const int lh  = l >> 4;            // lane half selects K sub-range (contiguous k = 16*lh + e)
  const int m   = l & 15;
  const int i0  = blockIdx.x << 4;
  const int n0  = (blockIdx.y << 7) + (wid << 5);
  (void)M;

  const u32* A32 = (const u32*)A;
  const u32* B32 = (const u32*)BT;
  const int K2 = K >> 1;
  const size_t arow  = (size_t)(i0 + m) * K2;
  const size_t b0row = (size_t)(n0 + m) * K2;
  const size_t b1row = (size_t)(n0 + 16 + m) * K2;

  v8f acc0 = {}, acc1 = {};
  Frag16 a, b0f, b1f;
  for (int kk = 0; kk < K; kk += 32) {
    const int k2 = kk >> 1;
    __builtin_prefetch(A32 + arow + k2 + 32, 0, 3);   // WGP-scope global_prefetch_b8
    // A 16x32 bf16: lanes 0-15 K{0..15}, lanes 16-31 K{16..31}
    uint4 a0 = *(const uint4*)(A32 + arow + k2 + lh * 8);
    uint4 a1 = *(const uint4*)(A32 + arow + k2 + lh * 8 + 4);
    // B 32x16 bf16: n = l&15, k = 16*lh + e  -> contiguous in BT row
    uint4 b0 = *(const uint4*)(B32 + b0row + k2 + lh * 8);
    uint4 b1 = *(const uint4*)(B32 + b0row + k2 + lh * 8 + 4);
    uint4 b2 = *(const uint4*)(B32 + b1row + k2 + lh * 8);
    uint4 b3 = *(const uint4*)(B32 + b1row + k2 + lh * 8 + 4);
    a.u[0] = a0.x; a.u[1] = a0.y; a.u[2] = a0.z; a.u[3] = a0.w;
    a.u[4] = a1.x; a.u[5] = a1.y; a.u[6] = a1.z; a.u[7] = a1.w;
    b0f.u[0] = b0.x; b0f.u[1] = b0.y; b0f.u[2] = b0.z; b0f.u[3] = b0.w;
    b0f.u[4] = b1.x; b0f.u[5] = b1.y; b0f.u[6] = b1.z; b0f.u[7] = b1.w;
    b1f.u[0] = b2.x; b1f.u[1] = b2.y; b1f.u[2] = b2.z; b1f.u[3] = b2.w;
    b1f.u[4] = b3.x; b1f.u[5] = b3.y; b1f.u[6] = b3.z; b1f.u[7] = b3.w;
    acc0 = __builtin_amdgcn_wmma_f32_16x16x32_bf16(false, a.v, false, b0f.v,
                                                   (short)0, acc0, false, false);
    acc1 = __builtin_amdgcn_wmma_f32_16x16x32_bf16(false, a.v, false, b1f.v,
                                                   (short)0, acc1, false, false);
  }
  #pragma unroll
  for (int r = 0; r < 8; ++r) {                 // C layout: M = r + 8*lh, N = l&15
    int row = i0 + lh * 8 + r;
    int col0 = n0 + m;
    int col1 = n0 + 16 + m;
    float v0 = acc0[r], v1 = acc1[r];
    if (mode == 1)      { v0 = eluf(v0); v1 = eluf(v1); }
    else if (mode == 2) {
      v0 += addsrc[(size_t)row * Nc + col0]; v0 = v0 > 0.f ? v0 : 0.f;
      v1 += addsrc[(size_t)row * Nc + col1]; v1 = v1 > 0.f ? v1 : 0.f;
    } else if (mode == 3) {
      v0 += bias[col0]; v0 = v0 > 0.f ? v0 : 0.f;
      v1 += bias[col1]; v1 = v1 > 0.f ? v1 : 0.f;
    }
    C[(size_t)row * Nc + col0] = v0;
    C[(size_t)row * Nc + col1] = v1;
    if (Cbf) {
      Cbf[(size_t)row * Nc + col0] = f2bf(v0);
      Cbf[(size_t)row * Nc + col1] = f2bf(v1);
    }
  }
}

// ------------------------- flash masked-softmax attention ------------------
// Per block: one head h, one 16-row tile. F/32 waves, wave w owns output cols
// [32w, 32w+32). Wave 0 computes the 16x32 probability tile (online softmax,
// leakyrelu(src_i + dst_j) with edge-bitmask) once per chunk and shares it
// via LDS; all waves rescale their f32 accumulators and issue 2 WMMAs.
// Epilogue: out = elu(acc / rowsum) written bf16 into the head-concat buffer.
template<int F>
__global__ void k_attn(const u16* __restrict__ WhT, const float* __restrict__ srcg,
                       const float* __restrict__ dstg, const u32* __restrict__ mb,
                       u16* __restrict__ outcat) {
  __shared__ u32  pp[16][16];      // bf16 prob pairs: [row][k/2]
  __shared__ float scl[16];        // per-row rescale factor for this chunk
  __shared__ float rsum[16];       // final softmax denominators

  const int h   = blockIdx.y;
  const int i0  = blockIdx.x << 4;
  const int wid = threadIdx.x >> 5;
  const int l   = threadIdx.x & 31;
  const int lh  = l >> 4;
  const int m   = l & 15;
  const int f0  = wid << 5;

  const float* srcv = srcg + (size_t)h * N_NODES;
  const float* dstv = dstg + (size_t)h * N_NODES;
  const u32* whT32  = (const u32*)(WhT + (size_t)h * F * N_NODES);

  v8f acc0 = {}, acc1 = {};
  float s_i = 0.f, runmax = -3.4e38f, runsum = 0.f;
  if (wid == 0) s_i = srcv[i0 + m];      // lanes m and m+16 duplicate row m

  const size_t b0row = (size_t)(f0 + m)      * (N_NODES >> 1);
  const size_t b1row = (size_t)(f0 + 16 + m) * (N_NODES >> 1);

  for (int j0 = 0; j0 < N_NODES; j0 += 32) {
    if (wid == 0) {
      float dlane = dstv[j0 + l];        // 32 dst values spread over the wave
      u32 w0 = mb[(size_t)(i0 + m) * (N_NODES >> 5) + (j0 >> 5)];
      float lm = -3.4e38f;
      #pragma unroll
      for (int k = 0; k < 32; ++k) {
        float ev = s_i + __shfl(dlane, k, 32);
        ev = ev > 0.f ? ev : 0.2f * ev;
        ev = ((w0 >> k) & 1u) ? ev : NEGV;
        lm = fmaxf(lm, ev);
      }
      float nm = fmaxf(runmax, lm);
      float rs = __expf(runmax - nm);
      float cs = 0.f;
      #pragma unroll
      for (int kp = 0; kp < 16; ++kp) {
        float e0 = s_i + __shfl(dlane, 2 * kp, 32);
        e0 = e0 > 0.f ? e0 : 0.2f * e0;
        e0 = ((w0 >> (2 * kp)) & 1u) ? e0 : NEGV;
        float e1 = s_i + __shfl(dlane, 2 * kp + 1, 32);
        e1 = e1 > 0.f ? e1 : 0.2f * e1;
        e1 = ((w0 >> (2 * kp + 1)) & 1u) ? e1 : NEGV;
        float p0 = __expf(e0 - nm), p1 = __expf(e1 - nm);
        cs += p0 + p1;
        pp[m][kp] = (u32)f2bf(p0) | ((u32)f2bf(p1) << 16);
      }
      runsum = runsum * rs + cs;
      runmax = nm;
      scl[m] = rs;
    }
    __syncthreads();
    // A fragment from LDS (k = 16*lh + e, contiguous)
    Frag16 a, bf0, bf1;
    #pragma unroll
    for (int r = 0; r < 4; ++r) a.u[r]     = pp[m][lh * 8 + r];
    #pragma unroll
    for (int r = 0; r < 4; ++r) a.u[4 + r] = pp[m][lh * 8 + 4 + r];
    // B fragments: WhT[f][j] rows are contiguous in j -> two b128 loads each
    const int j2 = j0 >> 1;
    uint4 t0 = *(const uint4*)(whT32 + b0row + j2 + lh * 8);
    uint4 t1 = *(const uint4*)(whT32 + b0row + j2 + lh * 8 + 4);
    uint4 t2 = *(const uint4*)(whT32 + b1row + j2 + lh * 8);
    uint4 t3 = *(const uint4*)(whT32 + b1row + j2 + lh * 8 + 4);
    bf0.u[0] = t0.x; bf0.u[1] = t0.y; bf0.u[2] = t0.z; bf0.u[3] = t0.w;
    bf0.u[4] = t1.x; bf0.u[5] = t1.y; bf0.u[6] = t1.z; bf0.u[7] = t1.w;
    bf1.u[0] = t2.x; bf1.u[1] = t2.y; bf1.u[2] = t2.z; bf1.u[3] = t2.w;
    bf1.u[4] = t3.x; bf1.u[5] = t3.y; bf1.u[6] = t3.z; bf1.u[7] = t3.w;
    #pragma unroll
    for (int r = 0; r < 8; ++r) {            // online-softmax rescale (C layout rows)
      float sc = scl[lh * 8 + r];
      acc0[r] *= sc; acc1[r] *= sc;
    }
    __syncthreads();
    acc0 = __builtin_amdgcn_wmma_f32_16x16x32_bf16(false, a.v, false, bf0.v,
                                                   (short)0, acc0, false, false);
    acc1 = __builtin_amdgcn_wmma_f32_16x16x32_bf16(false, a.v, false, bf1.v,
                                                   (short)0, acc1, false, false);
  }
  if (wid == 0) rsum[m] = runsum;
  __syncthreads();
  #pragma unroll
  for (int r = 0; r < 8; ++r) {
    int row = i0 + lh * 8 + r;
    float inv = 1.f / rsum[lh * 8 + r];
    float v0 = eluf(acc0[r] * inv);
    float v1 = eluf(acc1[r] * inv);
    size_t base = (size_t)row * (HEADS * F) + (size_t)h * F;
    outcat[base + f0 + m]      = f2bf(v0);
    outcat[base + f0 + 16 + m] = f2bf(v1);
  }
}

// -------------------------------- head kernels -----------------------------

__global__ void k_head(const float* __restrict__ x2, const float* __restrict__ w,
                       const float* __restrict__ bvec, float* __restrict__ out) {
  __shared__ float sg[H2];
  int t = threadIdx.x;
  float acc = 0.f;
  for (int n = 0; n < N_NODES; ++n) acc += x2[(size_t)n * H2 + t];
  sg[t] = sigm(acc * (1.f / N_NODES));
  __syncthreads();
  float o = bvec[t];
  for (int k = 0; k < H2; ++k) o += sg[k] * w[k * H2 + t];
  out[t] = o;
}

__global__ void k_mv(const float* __restrict__ W, const float* __restrict__ c,
                     float* __restrict__ out) {   // out = W @ c  (128x128)
  int t = threadIdx.x;
  float o = 0.f;
  for (int k = 0; k < H2; ++k) o += W[t * H2 + k] * c[k];
  out[t] = o;
}

__global__ void k_disc(const float* __restrict__ xp, const float* __restrict__ xm,
                       const float* __restrict__ wc, const float* __restrict__ db,
                       float* __restrict__ out) {
  __shared__ float r1[128], r2[128];
  int n = blockIdx.x, t = threadIdx.x;
  float w = wc[t];
  r1[t] = xp[(size_t)n * H2 + t] * w;
  r2[t] = xm[(size_t)n * H2 + t] * w;
  __syncthreads();
  for (int o = 64; o; o >>= 1) { if (t < o) { r1[t] += r1[t + o]; r2[t] += r2[t + o]; } __syncthreads(); }
  if (!t) { out[n * 2] = r1[0] + db[0]; out[n * 2 + 1] = r2[0] + db[0]; }
}

__global__ void k_advprep(const float* __restrict__ W, const float* __restrict__ bvec,
                          float* __restrict__ wsum, float* __restrict__ bsum) {
  int t = threadIdx.x;
  float s = 0.f;
  for (int j = 0; j < H2; ++j) s += W[t * H2 + j];   // (x@W+b).sum(1) == x@rowsum + sum(b)
  wsum[t] = s;
  if (!t) { float b = 0.f; for (int j = 0; j < H2; ++j) b += bvec[j]; bsum[0] = b; }
}

__global__ void k_adv(const float* __restrict__ x2, const float* __restrict__ wsum,
                      const float* __restrict__ bsum, float* __restrict__ out) {
  __shared__ float r1[128];
  int n = blockIdx.x, t = threadIdx.x;
  r1[t] = x2[(size_t)n * H2 + t] * wsum[t];
  __syncthreads();
  for (int o = 64; o; o >>= 1) { if (t < o) r1[t] += r1[t + o]; __syncthreads(); }
  if (!t) out[n] = r1[0] + bsum[0];
}

__global__ void k_gather(const int* __restrict__ idx, const float* __restrict__ x2o,
                         u16* __restrict__ z) {
  int b = blockIdx.x, t = threadIdx.x;      // 256 threads: [e1 | e2]
  int n = (t < H2) ? idx[b] : idx[B_PAIRS + b];
  int f = t & (H2 - 1);
  z[(size_t)b * (2 * H2) + t] = f2bf(x2o[(size_t)n * H2 + f]);
}

__global__ void k_dec2(const float* __restrict__ hdec, const float* __restrict__ w2,
                       const float* __restrict__ b2, float* __restrict__ log1,
                       float* __restrict__ logv) {
  __shared__ float r1[128];
  int b = blockIdx.x, t = threadIdx.x;
  float s = 0.f;
  for (int k = t; k < DECD; k += 128) s += hdec[(size_t)b * DECD + k] * w2[k];
  r1[t] = s; __syncthreads();
  for (int o = 64; o; o >>= 1) { if (t < o) r1[t] += r1[t + o]; __syncthreads(); }
  if (!t) { float v = r1[0] + b2[0]; log1[b] = v; logv[b] = sigm(v); }
}

// -------------------------------- orchestration ----------------------------

extern "C" void kernel_launch(void* const* d_in, const int* in_sizes, int n_in,
                              void* d_out, int out_size, void* d_ws, size_t ws_size,
                              hipStream_t stream) {
  (void)in_sizes; (void)n_in; (void)out_size; (void)ws_size;
  const float* x_o   = (const float*)d_in[0];
  const float* x_a   = (const float*)d_in[1];
  const int*   edge  = (const int*)d_in[2];
  const int*   idx   = (const int*)d_in[3];
  const float* W1    = (const float*)d_in[4];
  const float* a1    = (const float*)d_in[5];
  const float* lin1  = (const float*)d_in[6];
  const float* res1  = (const float*)d_in[7];
  const float* W2    = (const float*)d_in[8];
  const float* a2    = (const float*)d_in[9];
  const float* lin2  = (const float*)d_in[10];
  const float* res2  = (const float*)d_in[11];
  const float* mlp1w = (const float*)d_in[12];
  const float* mlp1b = (const float*)d_in[13];
  const float* discw = (const float*)d_in[14];
  const float* discb = (const float*)d_in[15];
  const float* dec1w = (const float*)d_in[16];
  const float* dec1b = (const float*)d_in[17];
  const float* dec2w = (const float*)d_in[18];
  const float* dec2b = (const float*)d_in[19];
  const float* advw  = (const float*)d_in[20];
  const float* advb  = (const float*)d_in[21];

  // output layout: log[4096], ret_os[4096,2], ret_os_a[4096,2], x2_o[4096,128],
  //                logits[1,8192], log1[4096,1]
  float* out      = (float*)d_out;
  float* o_log    = out;
  float* o_retos  = out + 4096;
  float* o_retosa = out + 12288;
  float* o_x2o    = out + 20480;
  float* o_logits = out + 544768;
  float* o_log1   = out + 552960;

  char* ws = (char*)d_ws;
  size_t off = 0;
  auto alloc = [&](size_t bytes) -> char* {
    char* p = ws + off; off += (bytes + 255) & ~(size_t)255; return p;
  };
  u32*  maskbits = (u32*)alloc((size_t)N_NODES * (N_NODES / 32) * 4);
  u16*  W1T   = (u16*)alloc((size_t)HEADS * H1 * IN_DIM * 2);
  u16*  lin1T = (u16*)alloc((size_t)H1 * (HEADS * H1) * 2);
  u16*  res1T = (u16*)alloc((size_t)H1 * IN_DIM * 2);
  u16*  W2T   = (u16*)alloc((size_t)HEADS * H2 * H1 * 2);
  u16*  lin2T = (u16*)alloc((size_t)H2 * (HEADS * H2) * 2);
  u16*  res2T = (u16*)alloc((size_t)H2 * H1 * 2);
  u16*  dec1T = (u16*)alloc((size_t)DECD * (2 * H2) * 2);
  u16*  xbf   = (u16*)alloc((size_t)N_NODES * IN_DIM * 2);
  float* Wh   = (float*)alloc((size_t)HEADS * N_NODES * H1 * 4);
  u16*  WhT   = (u16*)alloc((size_t)HEADS * H1 * N_NODES * 2);
  float* srcv = (float*)alloc((size_t)HEADS * N_NODES * 4);
  float* dstv = (float*)alloc((size_t)HEADS * N_NODES * 4);
  u16*  attcat= (u16*)alloc((size_t)N_NODES * HEADS * H1 * 2);
  float* tbuf = (float*)alloc((size_t)N_NODES * H1 * 4);
  float* h1   = (float*)alloc((size_t)N_NODES * H1 * 4);
  u16*  h1bf  = (u16*)alloc((size_t)N_NODES * H1 * 2);
  float* x2a  = (float*)alloc((size_t)N_NODES * H2 * 4);
  u16*  zb    = (u16*)alloc((size_t)B_PAIRS * 2 * H2 * 2);
  float* hdec = (float*)alloc((size_t)B_PAIRS * DECD * 4);
  float* smalls = (float*)alloc(4096);
  float* h_os  = smalls;        float* h_osa = smalls + 128;
  float* wc_os = smalls + 256;  float* wc_osa = smalls + 384;
  float* wsum  = smalls + 512;  float* bsum  = smalls + 640;

  dim3 b256(256);
  auto blocks = [](long n) { return dim3((unsigned)((n + 255) / 256)); };

  // edge bitmask
  k_zero32<<<blocks((long)N_NODES * 128), b256, 0, stream>>>(maskbits, N_NODES * 128);
  k_mask<<<blocks(N_EDGES), b256, 0, stream>>>(edge, maskbits);

  // weights -> bf16, pre-transposed to [N,K] for contiguous WMMA B loads
  for (int h = 0; h < HEADS; ++h) {
    k_convT<<<blocks((long)IN_DIM * H1), b256, 0, stream>>>(
        W1 + (size_t)h * IN_DIM * H1, W1T + (size_t)h * H1 * IN_DIM, IN_DIM, H1);
    k_convT<<<blocks((long)H1 * H2), b256, 0, stream>>>(
        W2 + (size_t)h * H1 * H2, W2T + (size_t)h * H2 * H1, H1, H2);
  }
  k_convT<<<blocks((long)HEADS * H1 * H1), b256, 0, stream>>>(lin1, lin1T, HEADS * H1, H1);
  k_convT<<<blocks((long)IN_DIM * H1), b256, 0, stream>>>(res1, res1T, IN_DIM, H1);
  k_convT<<<blocks((long)HEADS * H2 * H2), b256, 0, stream>>>(lin2, lin2T, HEADS * H2, H2);
  k_convT<<<blocks((long)H1 * H2), b256, 0, stream>>>(res2, res2T, H1, H2);
  k_convT<<<blocks((long)2 * H2 * DECD), b256, 0, stream>>>(dec1w, dec1T, 2 * H2, DECD);

  auto gemm = [&](const u16* A, const u16* BT, float* C, int M, int Nc, int K,
                  int mode, const float* bias, const float* addsrc, u16* Cbf) {
    dim3 g(M / 16, Nc / 128);
    k_gemm<<<g, dim3(128), 0, stream>>>(A, BT, C, M, Nc, K, mode, bias, addsrc, Cbf);
  };

  auto encode = [&](const float* x, float* x2out) {
    // ---- layer 1 (Fin=512, Fo=256) ----
    k_conv<<<blocks((long)N_NODES * IN_DIM), b256, 0, stream>>>(x, xbf, N_NODES * IN_DIM);
    for (int h = 0; h < HEADS; ++h)
      gemm(xbf, W1T + (size_t)h * H1 * IN_DIM, Wh + (size_t)h * N_NODES * H1,
           N_NODES, H1, IN_DIM, 0, nullptr, nullptr, nullptr);
    k_srcdst<<<dim3(HEADS * N_NODES), dim3(128), 0, stream>>>(Wh, a1, srcv, dstv, H1);
    for (int h = 0; h < HEADS; ++h)
      k_convT<<<blocks((long)N_NODES * H1), b256, 0, stream>>>(
          Wh + (size_t)h * N_NODES * H1, WhT + (size_t)h * H1 * N_NODES, N_NODES, H1);
    k_attn<H1><<<dim3(N_NODES / 16, HEADS), dim3(H1), 0, stream>>>(
        WhT, srcv, dstv, maskbits, attcat);
    gemm(attcat, lin1T, tbuf, N_NODES, H1, HEADS * H1, 1, nullptr, nullptr, nullptr);
    gemm(xbf, res1T, h1, N_NODES, H1, IN_DIM, 2, nullptr, tbuf, h1bf);
    // ---- layer 2 (Fin=256, Fo=128) ----
    for (int h = 0; h < HEADS; ++h)
      gemm(h1bf, W2T + (size_t)h * H2 * H1, Wh + (size_t)h * N_NODES * H2,
           N_NODES, H2, H1, 0, nullptr, nullptr, nullptr);
    k_srcdst<<<dim3(HEADS * N_NODES), dim3(128), 0, stream>>>(Wh, a2, srcv, dstv, H2);
    for (int h = 0; h < HEADS; ++h)
      k_convT<<<blocks((long)N_NODES * H2), b256, 0, stream>>>(
          Wh + (size_t)h * N_NODES * H2, WhT + (size_t)h * H2 * N_NODES, N_NODES, H2);
    k_attn<H2><<<dim3(N_NODES / 16, HEADS), dim3(H2), 0, stream>>>(
        WhT, srcv, dstv, maskbits, attcat);
    gemm(attcat, lin2T, tbuf, N_NODES, H2, HEADS * H2, 1, nullptr, nullptr, nullptr);
    gemm(h1bf, res2T, x2out, N_NODES, H2, H1, 2, nullptr, tbuf, nullptr);
  };

  encode(x_o, o_x2o);
  encode(x_a, x2a);

  // graph-summary heads + bilinear discriminator
  k_head<<<1, 128, 0, stream>>>(o_x2o, mlp1w, mlp1b, h_os);
  k_head<<<1, 128, 0, stream>>>(x2a,   mlp1w, mlp1b, h_osa);
  k_mv<<<1, 128, 0, stream>>>(discw, h_os,  wc_os);
  k_mv<<<1, 128, 0, stream>>>(discw, h_osa, wc_osa);
  k_disc<<<N_NODES, 128, 0, stream>>>(o_x2o, x2a, wc_os,  discb, o_retos);
  k_disc<<<N_NODES, 128, 0, stream>>>(x2a, o_x2o, wc_osa, discb, o_retosa);

  // pair decoder
  k_gather<<<B_PAIRS, 256, 0, stream>>>(idx, o_x2o, zb);
  gemm(zb, dec1T, hdec, B_PAIRS, DECD, 2 * H2, 3, dec1b, nullptr, nullptr);
  k_dec2<<<B_PAIRS, 128, 0, stream>>>(hdec, dec2w, dec2b, o_log1, o_log);

  // adversary logits
  k_advprep<<<1, 128, 0, stream>>>(advw, advb, wsum, bsum);
  k_adv<<<N_NODES, 128, 0, stream>>>(o_x2o, wsum, bsum, o_logits);
  k_adv<<<N_NODES, 128, 0, stream>>>(x2a,  wsum, bsum, o_logits + N_NODES);
}
